// Upsampling_36962488549966
// MI455X (gfx1250) — compile-verified, matrix operand order checked by reading, not run on hardware
//
#include <hip/hip_runtime.h>
#include <hip/hip_bf16.h>
#include <math.h>

// ---------------- problem constants ----------------
#define B_    16
#define S_    1024
#define D_    512
#define H_    8
#define DH_   64
#define POOL_ 4
#define NCOL  (B_ * S_)          // 16384 activation columns

typedef __attribute__((ext_vector_type(16))) __bf16 v16bf;
typedef __attribute__((ext_vector_type(8)))  float  v8f;

union Frag {
  v16bf v;
  unsigned short u[16];
  uint4 q[2];
};

__device__ __forceinline__ unsigned short f2bf(float f) {
  unsigned int u = __float_as_uint(f);
  unsigned int r = u + 0x7FFFu + ((u >> 16) & 1u);   // round-to-nearest-even
  return (unsigned short)(r >> 16);
}

__device__ __forceinline__ v8f wmma_bf16(const Frag& a, const Frag& b, v8f c) {
  return __builtin_amdgcn_wmma_f32_16x16x32_bf16(false, a.v, false, b.v,
                                                 (short)0, c, false, false);
}

// ---------------- prep kernels ----------------
__global__ void k_f32_to_bf16(const float* __restrict__ src,
                              unsigned short* __restrict__ dst, int n) {
  for (int i = blockIdx.x * blockDim.x + threadIdx.x; i < n;
       i += gridDim.x * blockDim.x)
    dst[i] = f2bf(src[i]);
}

__global__ void k_tsb(const float* __restrict__ ts_emb,
                      const int* __restrict__ te, float* __restrict__ tsb) {
  int i = blockIdx.x * blockDim.x + threadIdx.x;      // B*D = 8192
  if (i >= B_ * D_) return;
  int b = i >> 9, e = i & 511;
  tsb[i] = ts_emb[te[b] * D_ + e];
}

// ------------- conv (+pe) + GELU + LN1 (over D) ------------
// one block per (b,s) column; 256 threads, 2 channels each
__global__ __launch_bounds__(256)
void k_conv_ln1(const float* __restrict__ ts, const float* __restrict__ conv_w,
                const float* __restrict__ conv_b, const float* __restrict__ pe,
                const float* __restrict__ g, const float* __restrict__ be,
                float* __restrict__ convBuf,          // f32 (b,e,s)  residual
                unsigned short* __restrict__ n1t)     // bf16 (col,e)
{
  __shared__ float red[256];
  __shared__ float shx[POOL_];
  const int bs = blockIdx.x;
  const int b = bs >> 10, s = bs & 1023;
  const int t = threadIdx.x;
  if (t < POOL_) shx[t] = ts[b * 4096 + s * POOL_ + t];
  __syncthreads();

  float x[2];
  float sum = 0.f;
  for (int r = 0; r < 2; ++r) {
    int d = t + r * 256;
    float a = conv_b[d] + pe[s * D_ + d];
    for (int p = 0; p < POOL_; ++p) a += shx[p] * conv_w[d * POOL_ + p];
    convBuf[(b * D_ + d) * S_ + s] = a;               // pre-activation residual
    float gel = 0.5f * a * (1.0f + erff(a * 0.7071067811865475f));
    x[r] = gel;
    sum += gel;
  }
  red[t] = sum; __syncthreads();
  for (int o = 128; o > 0; o >>= 1) {
    if (t < o) red[t] += red[t + o];
    __syncthreads();
  }
  float mean = red[0] * (1.0f / D_);
  __syncthreads();
  float sq = 0.f;
  for (int r = 0; r < 2; ++r) { float d = x[r] - mean; sq += d * d; }
  red[t] = sq; __syncthreads();
  for (int o = 128; o > 0; o >>= 1) {
    if (t < o) red[t] += red[t + o];
    __syncthreads();
  }
  float rstd = rsqrtf(1e-5f + red[0] * (1.0f / (D_ - 1)));   // ddof=1
  float gs = g[s], bsft = be[s];
  for (int r = 0; r < 2; ++r) {
    int d = t + r * 256;
    n1t[bs * D_ + d] = f2bf(gs * (x[r] - mean) * rstd + bsft);
  }
}

// ------------- generic WMMA bf16 GEMM: C(512,NCOL) = W(512,512) * X -------------
// W bf16 row-major (M,K); Xt bf16 column-major per col: Xt[col*512 + k]
__global__ __launch_bounds__(256)
void k_gemm(const unsigned short* __restrict__ W,
            const unsigned short* __restrict__ Xt,
            const float* __restrict__ biasE,      // (512)  or null
            const float* __restrict__ biasBE,     // (B,512) or null
            const float* __restrict__ resBES,     // f32 (b,e,s) add, or null
            float* __restrict__ outF,             // f32 (b,e,s) or null
            unsigned short* __restrict__ outT,    // bf16 (col,e) or null
            unsigned short* __restrict__ outN)    // bf16 (b,e,s) or null
{
  const int tid  = threadIdx.x;
  const int lane = tid & 31;
  const int wid  = tid >> 5;
  const int wm   = wid & 3;
  const int wn   = wid >> 2;
  const int mbase = blockIdx.x * 128 + wm * 32;
  const int nbase = blockIdx.y * 64  + wn * 32;
  const int l15 = lane & 15;
  const int kb  = (lane >> 4) * 8;
  const int roff = (lane >> 4) * 8;

  v8f acc[2][2] = {};
  for (int k0 = 0; k0 < D_; k0 += 32) {
    Frag a0, a1, b0, b1;
    const unsigned short* ap0 = W + (mbase +      l15) * D_ + k0 + kb;
    const unsigned short* ap1 = W + (mbase + 16 + l15) * D_ + k0 + kb;
    a0.q[0] = *(const uint4*)(ap0); a0.q[1] = *(const uint4*)(ap0 + 16);
    a1.q[0] = *(const uint4*)(ap1); a1.q[1] = *(const uint4*)(ap1 + 16);
    const unsigned short* bp0 = Xt + (nbase +      l15) * D_ + k0 + kb;
    const unsigned short* bp1 = Xt + (nbase + 16 + l15) * D_ + k0 + kb;
    b0.q[0] = *(const uint4*)(bp0); b0.q[1] = *(const uint4*)(bp0 + 16);
    b1.q[0] = *(const uint4*)(bp1); b1.q[1] = *(const uint4*)(bp1 + 16);
    __builtin_prefetch(bp0 + 32, 0, 0);            // global_prefetch_b8
    acc[0][0] = wmma_bf16(a0, b0, acc[0][0]);
    acc[0][1] = wmma_bf16(a0, b1, acc[0][1]);
    acc[1][0] = wmma_bf16(a1, b0, acc[1][0]);
    acc[1][1] = wmma_bf16(a1, b1, acc[1][1]);
  }

  for (int ni = 0; ni < 2; ++ni) {
    int col = nbase + ni * 16 + l15;
    int b = col >> 10, s = col & 1023;
    for (int mi = 0; mi < 2; ++mi) {
      int e0 = mbase + mi * 16 + roff;
      float vals[8];
      for (int v = 0; v < 8; ++v) {
        int e = e0 + v;
        float x = acc[mi][ni][v];
        if (biasE)  x += biasE[e];
        if (biasBE) x += biasBE[b * D_ + e];
        if (resBES) x += resBES[(b * D_ + e) * S_ + s];
        vals[v] = x;
        if (outF) outF[(b * D_ + e) * S_ + s] = x;
        if (outN) outN[(b * D_ + e) * S_ + s] = f2bf(x);
      }
      if (outT) {
        unsigned short tu[8];
        for (int v = 0; v < 8; ++v) tu[v] = f2bf(vals[v]);
        uint4 pk;
        pk.x = (unsigned)tu[0] | ((unsigned)tu[1] << 16);
        pk.y = (unsigned)tu[2] | ((unsigned)tu[3] << 16);
        pk.z = (unsigned)tu[4] | ((unsigned)tu[5] << 16);
        pk.w = (unsigned)tu[6] | ((unsigned)tu[7] << 16);
        *(uint4*)(outT + col * D_ + e0) = pk;
      }
    }
  }
}

// ------------- flash attention (softmax over i; column j attends i<=j) -------------
// Qt,Kt bf16 (col=b*S+s, e=h*64+c); Vn bf16 (b,e,s) (padded +64); At bf16 (col,e)
__global__ __launch_bounds__(256)
void k_attn(const unsigned short* __restrict__ Qt,
            const unsigned short* __restrict__ Kt,
            const unsigned short* __restrict__ Vn,
            unsigned short* __restrict__ At)
{
  const int tid  = threadIdx.x;
  const int lane = tid & 31;
  const int wid  = tid >> 5;
  const int jbase = blockIdx.x * 128 + wid * 16;
  const int h = blockIdx.y, b = blockIdx.z;
  const int l15 = lane & 15;
  const int kb  = (lane >> 4) * 8;
  const int roff = (lane >> 4) * 8;
  const int j = jbase + l15;

  // B-fragments of scores: columns of K (loop invariant)
  Frag bk0, bk1;
  const unsigned short* kp = Kt + (b * S_ + j) * D_ + h * DH_ + kb;
  bk0.q[0] = *(const uint4*)(kp);      bk0.q[1] = *(const uint4*)(kp + 16);
  bk1.q[0] = *(const uint4*)(kp + 32); bk1.q[1] = *(const uint4*)(kp + 48);

  v8f acc[4] = {};
  float Mrun = -INFINITY, Lrun = 0.f;
  const int jhi = jbase + 15;

  for (int i0 = 0; i0 <= jhi; i0 += 32) {
    int r0 = i0 + l15;       if (r0 > S_ - 1) r0 = S_ - 1;   // masked anyway
    int r1 = i0 + 16 + l15;  if (r1 > S_ - 1) r1 = S_ - 1;
    Frag a00, a01, a10, a11;
    const unsigned short* qp0 = Qt + (b * S_ + r0) * D_ + h * DH_ + kb;
    a00.q[0] = *(const uint4*)(qp0);      a00.q[1] = *(const uint4*)(qp0 + 16);
    a01.q[0] = *(const uint4*)(qp0 + 32); a01.q[1] = *(const uint4*)(qp0 + 48);
    const unsigned short* qp1 = Qt + (b * S_ + r1) * D_ + h * DH_ + kb;
    a10.q[0] = *(const uint4*)(qp1);      a10.q[1] = *(const uint4*)(qp1 + 16);
    a11.q[0] = *(const uint4*)(qp1 + 32); a11.q[1] = *(const uint4*)(qp1 + 48);

    v8f d0 = {}, d1 = {};
    d0 = wmma_bf16(a00, bk0, d0); d0 = wmma_bf16(a01, bk1, d0);
    d1 = wmma_bf16(a10, bk0, d1); d1 = wmma_bf16(a11, bk1, d1);

    float p0[8], p1[8];
    float mx = -INFINITY;
    for (int v = 0; v < 8; ++v) {
      int ig0 = i0 + roff + v;
      int ig1 = i0 + 16 + roff + v;
      float s0 = (ig0 <= j) ? d0[v] * 0.125f : -INFINITY;   // 1/sqrt(64)
      float s1 = (ig1 <= j) ? d1[v] * 0.125f : -INFINITY;
      p0[v] = s0; p1[v] = s1;
      mx = fmaxf(mx, fmaxf(s0, s1));
    }
    mx = fmaxf(mx, __shfl_xor(mx, 16, 32));          // combine column halves
    float newM = fmaxf(Mrun, mx);
    float corr = __expf(Mrun - newM);
    float lsum = 0.f;
    for (int v = 0; v < 8; ++v) {
      p0[v] = __expf(p0[v] - newM);
      p1[v] = __expf(p1[v] - newM);
      lsum += p0[v] + p1[v];
    }
    lsum += __shfl_xor(lsum, 16, 32);
    Lrun = Lrun * corr + lsum;
    Mrun = newM;
    for (int mt = 0; mt < 4; ++mt) acc[mt] *= corr;

    // P -> B-fragment: lane-local (e<8 -> tile0 row e; e>=8 -> tile1 row e-8)
    Frag pb;
    for (int e = 0; e < 8; ++e)  pb.u[e]     = f2bf(p0[e]);
    for (int e = 0; e < 8; ++e)  pb.u[e + 8] = f2bf(p1[e]);

    for (int mt = 0; mt < 4; ++mt) {
      Frag va;
      const unsigned short* vp =
          Vn + ((size_t)(b * D_ + h * DH_ + mt * 16 + l15)) * S_ + i0 + kb;
      va.q[0] = *(const uint4*)(vp);
      va.q[1] = *(const uint4*)(vp + 16);   // may touch +64 pad at tensor edge
      acc[mt] = wmma_bf16(va, pb, acc[mt]);
    }
  }

  float invL = 1.0f / Lrun;
  for (int mt = 0; mt < 4; ++mt) {
    unsigned short tu[8];
    for (int v = 0; v < 8; ++v) tu[v] = f2bf(acc[mt][v] * invL);
    uint4 pk;
    pk.x = (unsigned)tu[0] | ((unsigned)tu[1] << 16);
    pk.y = (unsigned)tu[2] | ((unsigned)tu[3] << 16);
    pk.z = (unsigned)tu[4] | ((unsigned)tu[5] << 16);
    pk.w = (unsigned)tu[6] | ((unsigned)tu[7] << 16);
    *(uint4*)(At + (b * S_ + j) * D_ + h * DH_ + mt * 16 + roff) = pk;
  }
}

// ------------- LN2 over e for each (b,s) -------------
__global__ __launch_bounds__(256)
void k_ln2(const float* __restrict__ in, const float* __restrict__ g,
           const float* __restrict__ be, float* __restrict__ out)
{
  __shared__ float red[256];
  const int bs = blockIdx.x;
  const int b = bs >> 10, s = bs & 1023;
  const int t = threadIdx.x;
  float x[2];
  float sum = 0.f;
  for (int r = 0; r < 2; ++r) {
    int d = t + r * 256;
    x[r] = in[(b * D_ + d) * S_ + s];
    sum += x[r];
  }
  red[t] = sum; __syncthreads();
  for (int o = 128; o > 0; o >>= 1) { if (t < o) red[t] += red[t + o]; __syncthreads(); }
  float mean = red[0] * (1.0f / D_);
  __syncthreads();
  float sq = 0.f;
  for (int r = 0; r < 2; ++r) { float d = x[r] - mean; sq += d * d; }
  red[t] = sq; __syncthreads();
  for (int o = 128; o > 0; o >>= 1) { if (t < o) red[t] += red[t + o]; __syncthreads(); }
  float rstd = rsqrtf(1e-5f + red[0] * (1.0f / (D_ - 1)));
  float gs = g[s], bsft = be[s];
  for (int r = 0; r < 2; ++r) {
    int d = t + r * 256;
    out[(b * D_ + d) * S_ + s] = gs * (x[r] - mean) * rstd + bsft;
  }
}

// ---------------- host launcher ----------------
extern "C" void kernel_launch(void* const* d_in, const int* in_sizes, int n_in,
                              void* d_out, int out_size, void* d_ws, size_t ws_size,
                              hipStream_t stream) {
  const float* ts      = (const float*)d_in[0];
  const int*   te      = (const int*)  d_in[1];
  const float* conv_w  = (const float*)d_in[2];
  const float* conv_b  = (const float*)d_in[3];
  const float* pe_emb  = (const float*)d_in[4];
  const float* ts_emb  = (const float*)d_in[5];
  const float* ln1_g   = (const float*)d_in[6];
  const float* ln1_b   = (const float*)d_in[7];
  const float* dense_M = (const float*)d_in[8];
  const float* Wq      = (const float*)d_in[9];
  const float* Wk      = (const float*)d_in[10];
  const float* Wv      = (const float*)d_in[11];
  const float* Wo      = (const float*)d_in[12];
  const float* bo      = (const float*)d_in[13];
  const float* ln2_g   = (const float*)d_in[14];
  const float* ln2_b   = (const float*)d_in[15];
  (void)in_sizes; (void)n_in; (void)out_size; (void)ws_size;

  char* w = (char*)d_ws;
  size_t off = 0;
  auto take = [&](size_t bytes) -> void* {
    void* p = w + off;
    off += (bytes + 255) & ~(size_t)255;
    return p;
  };
  const size_t WB = (size_t)D_ * D_ * sizeof(unsigned short);
  unsigned short* Wd_b = (unsigned short*)take(WB);
  unsigned short* Wq_b = (unsigned short*)take(WB);
  unsigned short* Wk_b = (unsigned short*)take(WB);
  unsigned short* Wv_b = (unsigned short*)take(WB);
  unsigned short* Wo_b = (unsigned short*)take(WB);
  float* tsb           = (float*)take((size_t)B_ * D_ * sizeof(float));
  float* convBuf       = (float*)take((size_t)B_ * D_ * S_ * sizeof(float));
  unsigned short* n1t  = (unsigned short*)take((size_t)NCOL * D_ * 2);
  unsigned short* denT = (unsigned short*)take((size_t)NCOL * D_ * 2);
  unsigned short* Qt   = (unsigned short*)take((size_t)NCOL * D_ * 2);
  unsigned short* Kt   = (unsigned short*)take((size_t)NCOL * D_ * 2);
  unsigned short* Vn   = (unsigned short*)take((size_t)NCOL * D_ * 2 + 128); // +pad
  unsigned short* At   = (unsigned short*)take((size_t)NCOL * D_ * 2);
  float* resBuf        = (float*)take((size_t)B_ * D_ * S_ * sizeof(float));

  const int NW = D_ * D_;
  k_f32_to_bf16<<<256, 256, 0, stream>>>(dense_M, Wd_b, NW);
  k_f32_to_bf16<<<256, 256, 0, stream>>>(Wq, Wq_b, NW);
  k_f32_to_bf16<<<256, 256, 0, stream>>>(Wk, Wk_b, NW);
  k_f32_to_bf16<<<256, 256, 0, stream>>>(Wv, Wv_b, NW);
  k_f32_to_bf16<<<256, 256, 0, stream>>>(Wo, Wo_b, NW);
  k_tsb<<<(B_ * D_ + 255) / 256, 256, 0, stream>>>(ts_emb, te, tsb);

  k_conv_ln1<<<NCOL, 256, 0, stream>>>(ts, conv_w, conv_b, pe_emb,
                                       ln1_g, ln1_b, convBuf, n1t);

  dim3 gg(D_ / 128, NCOL / 64);
  // dense = M @ n1 + ts_emb[te]  -> bf16 transposed
  k_gemm<<<gg, 256, 0, stream>>>(Wd_b, n1t, nullptr, tsb, nullptr,
                                 nullptr, denT, nullptr);
  // Q, K (transposed bf16), V ((b,e,s) bf16)
  k_gemm<<<gg, 256, 0, stream>>>(Wq_b, denT, nullptr, nullptr, nullptr,
                                 nullptr, Qt, nullptr);
  k_gemm<<<gg, 256, 0, stream>>>(Wk_b, denT, nullptr, nullptr, nullptr,
                                 nullptr, Kt, nullptr);
  k_gemm<<<gg, 256, 0, stream>>>(Wv_b, denT, nullptr, nullptr, nullptr,
                                 nullptr, nullptr, Vn);

  dim3 ga(S_ / 128, H_, B_);
  k_attn<<<ga, 256, 0, stream>>>(Qt, Kt, Vn, At);

  // out = Wo @ cat + bo + conv (residual) -> f32
  k_gemm<<<gg, 256, 0, stream>>>(Wo_b, At, bo, nullptr, convBuf,
                                 resBuf, nullptr, nullptr);

  k_ln2<<<NCOL, 256, 0, stream>>>(resBuf, ln2_g, ln2_b, (float*)d_out);
}
